// StatTransformer_4080218931573
// MI455X (gfx1250) — matrix operand order repaired and checked by reading, not compiled, not run on hardware
//
#include <hip/hip_runtime.h>
#include <hip/hip_bf16.h>
#include <stdint.h>

// Model dims
#define Vv  32000
#define Dm  1024
#define Hh  16
#define Ll  4
#define DHd 64
#define DFF 4096
#define Bb  2
#define Ss  2048
#define Tt  (Bb*Ss)

typedef __attribute__((ext_vector_type(16))) __bf16 bf16x16;
typedef __attribute__((ext_vector_type(8)))  __bf16 bf16x8;
typedef __attribute__((ext_vector_type(4)))  __bf16 bf16x4;
typedef __attribute__((ext_vector_type(8)))  float  f32x8;

// ---------------------------------------------------------------------------
// WMMA helpers
// ---------------------------------------------------------------------------
__device__ __forceinline__ f32x8 wmma_bf16(bf16x16 a, bf16x16 b, f32x8 c) {
  return __builtin_amdgcn_wmma_f32_16x16x32_bf16(
      /*neg_a=*/false, a, /*neg_b=*/false, b,
      /*c_mod=*/(short)0, c, /*reuse_a=*/false, /*reuse_b=*/false);
}

// A-operand (16x32, 16-bit): lane holds row (lane&15); lane<16 -> K 0..15,
// lane>=16 -> K 16..31 (contiguous, matches B). Two 16B loads.
__device__ __forceinline__ bf16x16 load_fragA(const __bf16* base, int ld,
                                              int row, int kbase, int lane) {
  const __bf16* p = base + (size_t)(row + (lane & 15)) * ld + kbase + ((lane >> 4) << 4);
  bf16x8 lo = *(const bf16x8*)p;          // K = koff .. koff+7
  bf16x8 hi = *(const bf16x8*)(p + 8);    // K = koff+8 .. koff+15
  bf16x16 f;
#pragma unroll
  for (int i = 0; i < 8; ++i) { f[i] = lo[i]; f[8 + i] = hi[i]; }
  return f;
}

// B-operand (32x16, 16-bit): lane holds column (lane&15); lanes 0-15 K=0..15,
// lanes 16-31 K=16..31 (contiguous). One 32B contiguous load (two bf16x8).
// For C = A * W^T with W row-major [N,K], "column n" == row n of W.
__device__ __forceinline__ bf16x16 load_fragB(const __bf16* base, int ld,
                                              int col, int kbase, int lane) {
  const __bf16* p = base + (size_t)(col + (lane & 15)) * ld + kbase + ((lane >> 4) << 4);
  bf16x8 lo = *(const bf16x8*)p;
  bf16x8 hi = *(const bf16x8*)(p + 8);
  bf16x16 f;
#pragma unroll
  for (int i = 0; i < 8; ++i) { f[i] = lo[i]; f[8 + i] = hi[i]; }
  return f;
}

// ---------------------------------------------------------------------------
// f32 -> bf16 conversion (grid-stride, 4 elems/thread)
// ---------------------------------------------------------------------------
__global__ void __launch_bounds__(256)
cvt_f32_bf16_kernel(const float* __restrict__ s, __bf16* __restrict__ d, long long n4) {
  long long i = (long long)blockIdx.x * blockDim.x + threadIdx.x;
  long long stride = (long long)gridDim.x * blockDim.x;
  for (; i < n4; i += stride) {
    float4 v = ((const float4*)s)[i];
    bf16x4 o;
    o[0] = (__bf16)v.x; o[1] = (__bf16)v.y; o[2] = (__bf16)v.z; o[3] = (__bf16)v.w;
    ((bf16x4*)d)[i] = o;
  }
}

// ---------------------------------------------------------------------------
// Embedding gather + positional decay + pad mask -> x (f32) and x (bf16)
// ---------------------------------------------------------------------------
__global__ void __launch_bounds__(256)
embed_kernel(const int* __restrict__ ids, const unsigned char* __restrict__ pad,
             const float* __restrict__ emb, float* __restrict__ x,
             __bf16* __restrict__ xb) {
  int idx = blockIdx.x * blockDim.x + threadIdx.x;   // Tt * Dm/4 threads
  int t = idx >> 8;                                  // Dm/4 == 256
  int j = (idx & 255) << 2;
  int s = t & (Ss - 1);
  float f = pad[t] ? __expf(-0.1f * (float)(Ss - 1 - s)) : 0.f;
  int id = ids[t];
  float4 e = *(const float4*)(emb + (size_t)id * Dm + j);
  float4 v = { e.x * f, e.y * f, e.z * f, e.w * f };
  *(float4*)(x + (size_t)t * Dm + j) = v;
  bf16x4 o;
  o[0] = (__bf16)v.x; o[1] = (__bf16)v.y; o[2] = (__bf16)v.z; o[3] = (__bf16)v.w;
  *((bf16x4*)(xb + (size_t)t * Dm + j)) = o;
}

// ---------------------------------------------------------------------------
// WMMA GEMM: C[M,N] = act(A[M,K](bf16) @ W[N,K]^T(bf16)). One wave = 32x32 tile.
// ---------------------------------------------------------------------------
__global__ void __launch_bounds__(128)
gemm_wmma_kernel(const __bf16* __restrict__ A, const __bf16* __restrict__ W,
                 float* __restrict__ outF, __bf16* __restrict__ outB,
                 int M, int N, int K, int relu) {
  const int lane = threadIdx.x & 31;
  const int wave = threadIdx.x >> 5;
  const int tilesN = N >> 5;
  const int tiles  = (M >> 5) * tilesN;
  const int tile = blockIdx.x * 4 + wave;
  if (tile >= tiles) return;
  const int tm = (tile / tilesN) << 5;
  const int tn = (tile % tilesN) << 5;

  f32x8 c00 = {}, c01 = {}, c10 = {}, c11 = {};
  for (int k = 0; k < K; k += 32) {
    if (k + 32 < K) {  // -> global_prefetch_b8
      __builtin_prefetch(A + (size_t)(tm + (lane & 15)) * K + k + 32, 0, 1);
      __builtin_prefetch(W + (size_t)(tn + (lane & 15)) * K + k + 32, 0, 1);
    }
    bf16x16 a0 = load_fragA(A, K, tm,      k, lane);
    bf16x16 a1 = load_fragA(A, K, tm + 16, k, lane);
    bf16x16 b0 = load_fragB(W, K, tn,      k, lane);
    bf16x16 b1 = load_fragB(W, K, tn + 16, k, lane);
    c00 = wmma_bf16(a0, b0, c00);
    c01 = wmma_bf16(a0, b1, c01);
    c10 = wmma_bf16(a1, b0, c10);
    c11 = wmma_bf16(a1, b1, c11);
  }

  const int n0 = tn + (lane & 15);
  const int m0 = tm + ((lane >> 4) << 3);
  auto emit = [&](int m, int n, float v) {
    if (relu) v = fmaxf(v, 0.f);
    size_t idx = (size_t)m * N + n;
    if (outF) outF[idx] = v;
    if (outB) outB[idx] = (__bf16)v;
  };
#pragma unroll
  for (int r = 0; r < 8; ++r) {
    emit(m0 + r,      n0,      c00[r]);
    emit(m0 + r,      n0 + 16, c01[r]);
    emit(m0 + 16 + r, n0,      c10[r]);
    emit(m0 + 16 + r, n0 + 16, c11[r]);
  }
}

// ---------------------------------------------------------------------------
// Flash attention: one wave = (b, h, 16-row q tile); online softmax over
// 32-key steps; QK^T and P*V on WMMA; P transposed via LDS.
// ---------------------------------------------------------------------------
__global__ void __launch_bounds__(128)
attn_kernel(const __bf16* __restrict__ Q, const __bf16* __restrict__ Km,
            const __bf16* __restrict__ Vm, const unsigned char* __restrict__ pad,
            float* __restrict__ O) {
  __shared__ __bf16 plds[4][16 * 32];
  const int lane = threadIdx.x & 31;
  const int wave = threadIdx.x >> 5;
  const int gw = blockIdx.x * 4 + wave;     // Bb*Hh*(Ss/16) total waves
  const int QT = Ss / 16;                   // 128
  const int qt = gw % QT;
  const int bh = gw / QT;
  const int b  = bh >> 4;
  const int h  = bh & 15;
  const int qtok = b * Ss + qt * 16;
  const int hd = h * DHd;
  const float scl = 0.125f;                 // 1/sqrt(64)

  bf16x16 aq0 = load_fragA(Q + hd, Dm, qtok, 0,  lane);
  bf16x16 aq1 = load_fragA(Q + hd, Dm, qtok, 32, lane);

  float mrun[8], lrun[8];
  f32x8 o0 = {}, o1 = {}, o2 = {}, o3 = {};
#pragma unroll
  for (int r = 0; r < 8; ++r) { mrun[r] = -1e30f; lrun[r] = 0.f; }

  const int n   = lane & 15;
  const int mh  = (lane >> 4) << 3;   // C/D-layout row offset
  const int kbB = (lane >> 4) << 4;   // B-layout K offset
  const int nk  = ((qt * 16 + 16) + 31) & ~31;

  for (int kb0 = 0; kb0 < nk; kb0 += 32) {
    const int ktok = b * Ss + kb0;
    bf16x16 bk00 = load_fragB(Km + hd, Dm, ktok,      0,  lane);
    bf16x16 bk01 = load_fragB(Km + hd, Dm, ktok,      32, lane);
    bf16x16 bk10 = load_fragB(Km + hd, Dm, ktok + 16, 0,  lane);
    bf16x16 bk11 = load_fragB(Km + hd, Dm, ktok + 16, 32, lane);
    f32x8 s0 = {}, s1 = {};
    s0 = wmma_bf16(aq0, bk00, s0);
    s0 = wmma_bf16(aq1, bk01, s0);
    s1 = wmma_bf16(aq0, bk10, s1);
    s1 = wmma_bf16(aq1, bk11, s1);

    const int k0 = kb0 + n, k1 = kb0 + 16 + n;
    const bool v0ok = pad[b * Ss + k0] != 0;
    const bool v1ok = pad[b * Ss + k1] != 0;
    float p0[8], p1[8];
#pragma unroll
    for (int r = 0; r < 8; ++r) {
      int qpos = qt * 16 + mh + r;
      float e0 = (v0ok && k0 <= qpos) ? s0[r] * scl : -1e9f;
      float e1 = (v1ok && k1 <= qpos) ? s1[r] * scl : -1e9f;
      float mx = fmaxf(e0, e1);
      mx = fmaxf(mx, __shfl_xor(mx, 8, 16));
      mx = fmaxf(mx, __shfl_xor(mx, 4, 16));
      mx = fmaxf(mx, __shfl_xor(mx, 2, 16));
      mx = fmaxf(mx, __shfl_xor(mx, 1, 16));
      float mnew  = fmaxf(mrun[r], mx);
      float alpha = __expf(mrun[r] - mnew);
      mrun[r] = mnew;
      p0[r] = __expf(e0 - mnew);
      p1[r] = __expf(e1 - mnew);
      float rs = p0[r] + p1[r];
      rs += __shfl_xor(rs, 8, 16);
      rs += __shfl_xor(rs, 4, 16);
      rs += __shfl_xor(rs, 2, 16);
      rs += __shfl_xor(rs, 1, 16);
      lrun[r] = lrun[r] * alpha + rs;
      o0[r] *= alpha; o1[r] *= alpha; o2[r] *= alpha; o3[r] *= alpha;
    }

    // D-layout -> A-layout transpose of P through LDS (same-wave DS is in-order)
    __bf16* pl = plds[wave];
#pragma unroll
    for (int r = 0; r < 8; ++r) {
      pl[(mh + r) * 32 + n]      = (__bf16)p0[r];
      pl[(mh + r) * 32 + 16 + n] = (__bf16)p1[r];
    }
    bf16x16 pA = load_fragA(pl, 32, 0, 0, lane);

    // V operand: B[kk, n] = x[ktok+kk][hd + c*16 + n]; 16 strided gathers/lane
    const __bf16* vp = Vm + (size_t)ktok * Dm + hd;
#pragma unroll
    for (int c = 0; c < 4; ++c) {
      const __bf16* col = vp + c * 16 + n;
      bf16x16 bv;
#pragma unroll
      for (int j = 0; j < 16; ++j) bv[j] = col[(size_t)(kbB + j) * Dm];
      f32x8& oc = (c == 0) ? o0 : (c == 1) ? o1 : (c == 2) ? o2 : o3;
      oc = wmma_bf16(pA, bv, oc);
    }
  }

#pragma unroll
  for (int r = 0; r < 8; ++r) {
    float inv = 1.f / lrun[r];
    size_t base = (size_t)(qtok + mh + r) * Dm + hd + n;
    O[base]      = o0[r] * inv;
    O[base + 16] = o1[r] * inv;
    O[base + 32] = o2[r] * inv;
    O[base + 48] = o3[r] * inv;
  }
}

// ---------------------------------------------------------------------------
// x = LayerNorm(x + add) * g + b; optional bf16 mirror. One block per token.
// ---------------------------------------------------------------------------
__global__ void __launch_bounds__(256)
add_ln_kernel(float* __restrict__ x, const float* __restrict__ add,
              const float* __restrict__ g, const float* __restrict__ bt,
              __bf16* __restrict__ xb) {
  const int t = blockIdx.x;
  const int tid = threadIdx.x;
  const size_t base = (size_t)t * Dm + tid * 4;
  float4 v = *(const float4*)(x + base);
  if (add) {
    float4 a = *(const float4*)(add + base);
    v.x += a.x; v.y += a.y; v.z += a.z; v.w += a.w;
  }
  __shared__ float sm[8];
  __shared__ float stat[2];

  float s = v.x + v.y + v.z + v.w;
#pragma unroll
  for (int off = 16; off >= 1; off >>= 1) s += __shfl_xor(s, off, 32);
  if ((tid & 31) == 0) sm[tid >> 5] = s;
  __syncthreads();
  if (tid == 0) {
    float tot = 0.f;
    for (int i = 0; i < 8; ++i) tot += sm[i];
    stat[0] = tot * (1.0f / Dm);
  }
  __syncthreads();
  const float mu = stat[0];

  float dx0 = v.x - mu, dx1 = v.y - mu, dx2 = v.z - mu, dx3 = v.w - mu;
  float q = dx0 * dx0 + dx1 * dx1 + dx2 * dx2 + dx3 * dx3;
#pragma unroll
  for (int off = 16; off >= 1; off >>= 1) q += __shfl_xor(q, off, 32);
  __syncthreads();
  if ((tid & 31) == 0) sm[tid >> 5] = q;
  __syncthreads();
  if (tid == 0) {
    float tot = 0.f;
    for (int i = 0; i < 8; ++i) tot += sm[i];
    stat[1] = tot * (1.0f / Dm);
  }
  __syncthreads();
  const float inv = rsqrtf(stat[1] + 1e-5f);

  const int d = tid * 4;
  float y0 = dx0 * inv * g[d + 0] + bt[d + 0];
  float y1 = dx1 * inv * g[d + 1] + bt[d + 1];
  float y2 = dx2 * inv * g[d + 2] + bt[d + 2];
  float y3 = dx3 * inv * g[d + 3] + bt[d + 3];
  float4 o = { y0, y1, y2, y3 };
  *(float4*)(x + base) = o;
  if (xb) {
    bf16x4 ob;
    ob[0] = (__bf16)y0; ob[1] = (__bf16)y1; ob[2] = (__bf16)y2; ob[3] = (__bf16)y3;
    *((bf16x4*)(xb + base)) = ob;
  }
}

// ---------------------------------------------------------------------------
// logits[b, v] = dot(x_last[b], emb[v]) / 32   (M=2: plain VALU dot products)
// ---------------------------------------------------------------------------
__global__ void __launch_bounds__(256)
logits_kernel(const float* __restrict__ x, const float* __restrict__ emb,
              float* __restrict__ out) {
  int i = blockIdx.x * blockDim.x + threadIdx.x;
  if (i >= Bb * Vv) return;
  int b = i / Vv, v = i - b * Vv;
  const float* xr = x + (size_t)(b * Ss + Ss - 1) * Dm;
  const float* er = emb + (size_t)v * Dm;
  float acc = 0.f;
  for (int d = 0; d < Dm; d += 4) {
    float4 a = *(const float4*)(xr + d);
    float4 e = *(const float4*)(er + d);
    acc = fmaf(a.x, e.x, fmaf(a.y, e.y, fmaf(a.z, e.z, fmaf(a.w, e.w, acc))));
  }
  out[i] = acc * 0.03125f;
}

// ---------------------------------------------------------------------------
// Host launcher
// ---------------------------------------------------------------------------
extern "C" void kernel_launch(void* const* d_in, const int* in_sizes, int n_in,
                              void* d_out, int out_size, void* d_ws, size_t ws_size,
                              hipStream_t stream) {
  (void)in_sizes; (void)n_in; (void)out_size; (void)ws_size;
  const int*           ids  = (const int*)d_in[0];
  const unsigned char* pad  = (const unsigned char*)d_in[1];
  const float*         emb  = (const float*)d_in[2];
  const float*         Wq   = (const float*)d_in[3];
  const float*         Wk   = (const float*)d_in[4];
  const float*         W1   = (const float*)d_in[5];
  const float*         W2   = (const float*)d_in[6];
  const float*         g1   = (const float*)d_in[7];
  const float*         b1   = (const float*)d_in[8];
  const float*         g2   = (const float*)d_in[9];
  const float*         b2   = (const float*)d_in[10];
  const float*         outg = (const float*)d_in[11];
  const float*         outb = (const float*)d_in[12];
  float* logits = (float*)d_out;

  char* p = (char*)d_ws;
  auto alloc = [&](size_t bytes) -> void* {
    void* r = (void*)p;
    p += (bytes + 255) & ~(size_t)255;
    return r;
  };
  __bf16* wq_b = (__bf16*)alloc((size_t)Ll * Dm * Dm * 2);
  __bf16* wk_b = (__bf16*)alloc((size_t)Ll * Dm * Dm * 2);
  __bf16* w1_b = (__bf16*)alloc((size_t)Ll * DFF * Dm * 2);
  __bf16* w2_b = (__bf16*)alloc((size_t)Ll * Dm * DFF * 2);
  float*  xf   = (float*) alloc((size_t)Tt * Dm * 4);
  __bf16* xb   = (__bf16*)alloc((size_t)Tt * Dm * 2);
  __bf16* qb   = (__bf16*)alloc((size_t)Tt * Dm * 2);
  __bf16* kb   = (__bf16*)alloc((size_t)Tt * Dm * 2);
  float*  acc  = (float*) alloc((size_t)Tt * Dm * 4);
  __bf16* hb   = (__bf16*)alloc((size_t)Tt * DFF * 2);

  // Weight down-conversion (every call; deterministic)
  cvt_f32_bf16_kernel<<<2048, 256, 0, stream>>>(Wq, wq_b, (long long)Ll * Dm * Dm / 4);
  cvt_f32_bf16_kernel<<<2048, 256, 0, stream>>>(Wk, wk_b, (long long)Ll * Dm * Dm / 4);
  cvt_f32_bf16_kernel<<<2048, 256, 0, stream>>>(W1, w1_b, (long long)Ll * DFF * Dm / 4);
  cvt_f32_bf16_kernel<<<2048, 256, 0, stream>>>(W2, w2_b, (long long)Ll * Dm * DFF / 4);

  // Embedding + positional decay + pad
  embed_kernel<<<(Tt * Dm / 4) / 256, 256, 0, stream>>>(ids, pad, emb, xf, xb);

  const int gridQK   = ((Tt / 32) * (Dm / 32) + 3) / 4;   // 1024
  const int gridFFN1 = ((Tt / 32) * (DFF / 32) + 3) / 4;  // 4096
  const int gridFFN2 = ((Tt / 32) * (Dm / 32) + 3) / 4;   // 1024
  const int gridAttn = (Bb * Hh * (Ss / 16) + 3) / 4;     // 1024

  for (int l = 0; l < Ll; ++l) {
    const __bf16* wql = wq_b + (size_t)l * Dm * Dm;
    const __bf16* wkl = wk_b + (size_t)l * Dm * Dm;
    const __bf16* w1l = w1_b + (size_t)l * DFF * Dm;
    const __bf16* w2l = w2_b + (size_t)l * Dm * DFF;

    gemm_wmma_kernel<<<gridQK, 128, 0, stream>>>(xb, wql, nullptr, qb, Tt, Dm, Dm, 0);
    gemm_wmma_kernel<<<gridQK, 128, 0, stream>>>(xb, wkl, nullptr, kb, Tt, Dm, Dm, 0);
    attn_kernel<<<gridAttn, 128, 0, stream>>>(qb, kb, xb, pad, acc);
    add_ln_kernel<<<Tt, 256, 0, stream>>>(xf, acc, g1 + (size_t)l * Dm, b1 + (size_t)l * Dm, xb);

    gemm_wmma_kernel<<<gridFFN1, 128, 0, stream>>>(xb, w1l, nullptr, hb, Tt, DFF, Dm, 1);
    gemm_wmma_kernel<<<gridFFN2, 128, 0, stream>>>(hb, w2l, acc, nullptr, Tt, Dm, DFF, 0);
    add_ln_kernel<<<Tt, 256, 0, stream>>>(xf, acc, g2 + (size_t)l * Dm, b2 + (size_t)l * Dm, xb);
  }

  // Final LN + logits
  add_ln_kernel<<<Tt, 256, 0, stream>>>(xf, nullptr, outg, outb, nullptr);
  logits_kernel<<<(Bb * Vv + 255) / 256, 256, 0, stream>>>(xf, emb, logits);
}
